// InvariantPointAttentionV49_46961172414800
// MI455X (gfx1250) — compile-verified, matrix operand order checked by reading, not run on hardware
//
#include <hip/hip_runtime.h>
#include <hip/hip_bf16.h>

#define NTOK 1024
#define CDIM 384
#define HEADS 12
#define PPTS 4
#define DPAIR 128
#define CH 32
#define HP3 144            // H*P*3
#define FEAT 64            // 32 scalar + 12 point dims + pad
#define SCALE 0.17677669529663687f   // 32^-0.5
#define EPS 1e-5f
#define PBROWS 256         // pair rows per block in pair_bias

typedef __attribute__((ext_vector_type(16))) __bf16 v16bf;
typedef __attribute__((ext_vector_type(2)))  __bf16 v2bf;
typedef __attribute__((ext_vector_type(8)))  float v8f;

static __device__ __forceinline__ __bf16 f2bf(float f) {
    return (__bf16)f;     // native v_cvt_pk_bf16_f32 on gfx1250
}

// 16-bit A/B fragment K-offset for element j (0..15) of a lane (ISA 7.12.2):
// lanes 0-15 hold K {0..7, 16..23}; lanes 16-31 hold K {8..15, 24..31}
static __device__ __forceinline__ int frag_koff(int lane, int j) {
    return ((lane & 16) ? 8 : 0) + (j < 8 ? j : j + 8);
}

// inverse mapping: chunk offset c (0..31) -> (lane half, element j)
static __device__ __forceinline__ int inv_half(int c) { return (c >> 3) & 1; }
static __device__ __forceinline__ int inv_j(int c) { return (c & 7) + ((c >> 4) << 3); }

// CDNA5 async memory->LDS staging (ASYNCcnt-tracked, no VGPR round trip)
static __device__ __forceinline__ void async_ld_b128(void* lds_ptr, const void* gptr) {
    unsigned loff = (unsigned)(unsigned long long)lds_ptr;   // low 32 bits = LDS offset
    asm volatile("global_load_async_to_lds_b128 %0, %1, off"
                 :: "v"(loff), "v"((unsigned long long)gptr)
                 : "memory");
}
static __device__ __forceinline__ void wait_async0() {
    asm volatile("s_wait_asynccnt 0" ::: "memory");
}

// ---------------------------------------------------------------------------
// Generic one-wave 16x16-tile GEMM: D[M,Nout] = A[M,K] @ W[K,Nout] + bias
// ---------------------------------------------------------------------------
__global__ void gemm16_kernel(const float* __restrict__ A,
                              const float* __restrict__ W,
                              const float* __restrict__ bias,
                              float* __restrict__ D,
                              int M, int K, int Nout) {
    __shared__ float sA[16 * 32];
    __shared__ float sB[32 * 16];
    int lane = threadIdx.x;                   // 32 threads = 1 wave
    int m0 = blockIdx.x * 16;
    int n0 = blockIdx.y * 16;
    v8f c = {};
    int nk = (K + 31) / 32;
    int row = lane & 15;
    for (int kc = 0; kc < nk; ++kc) {
        int kb = kc * 32;
#pragma unroll
        for (int t0 = 0; t0 < 512; t0 += 32) {
            int t = t0 + lane;
            int r = t >> 5, col = t & 31;
            int kk = kb + col;
            int kcl = kk < K ? kk : K - 1;            // unconditional load + select
            float av = A[(size_t)(m0 + r) * K + kcl];
            sA[t] = kk < K ? av : 0.0f;
        }
#pragma unroll
        for (int t0 = 0; t0 < 512; t0 += 32) {
            int t = t0 + lane;
            int r = t >> 4, col = t & 15;
            int kk = kb + r;
            int kcl = kk < K ? kk : K - 1;
            float bv = W[(size_t)kcl * Nout + n0 + col];
            sB[t] = kk < K ? bv : 0.0f;
        }
        __syncthreads();
        v16bf a, b;
#pragma unroll
        for (int j = 0; j < 16; ++j) {
            int ko = frag_koff(lane, j);
            a[j] = f2bf(sA[row * 32 + ko]);
            b[j] = f2bf(sB[ko * 16 + row]);
        }
        c = __builtin_amdgcn_wmma_f32_16x16x32_bf16(false, a, false, b,
                                                    (short)0, c, false, false);
        __syncthreads();
    }
    int n = lane & 15;
    int mb = (lane & 16) ? 8 : 0;
#pragma unroll
    for (int r = 0; r < 8; ++r) {
        int m = mb + r;
        D[(size_t)(m0 + m) * Nout + n0 + n] = c[r] + bias[n0 + n];
    }
}

// ---------------------------------------------------------------------------
// Rotate/translate points to global frame; build:
//   qfeat[h][n][64]  f32  = [SCALE*q(32) | SCALE*q_pts(12) | 0]
//   kfragB            bf16, WMMA-B-fragment-ready: [h][kt][frag2][lane32][16]
//   vfragB            bf16, WMMA-B-fragment-ready: [h][kc][ct4][lane32][16]
//   rq[h][n] = -0.5*SCALE*|q_pts|^2    rk[h][n] = -0.5*SCALE*|k_pts|^2
// ---------------------------------------------------------------------------
__global__ void points_kernel(const float* __restrict__ q,
                              const float* __restrict__ k,
                              const float* __restrict__ v,
                              const float* __restrict__ qp,
                              const float* __restrict__ kp,
                              const float* __restrict__ vp,
                              const float* __restrict__ rot,
                              const float* __restrict__ trans,
                              float* __restrict__ qfeat,
                              __bf16* __restrict__ kfragB,
                              __bf16* __restrict__ vfragB,
                              float* __restrict__ rq,
                              float* __restrict__ rk) {
    int idx = blockIdx.x * blockDim.x + threadIdx.x;
    if (idx >= NTOK * HEADS) return;
    int n = idx / HEADS, h = idx % HEADS;
    const float* R = rot + n * 9;
    const float* t = trans + n * 3;
    float* qf = qfeat + ((size_t)h * NTOK + n) * FEAT;

    float kpt[12], vpt[12], qpt[12];
    float q2 = 0.0f, k2 = 0.0f;
#pragma unroll
    for (int p = 0; p < PPTS; ++p) {
        const float* sq = qp + (size_t)n * HP3 + h * 12 + p * 3;
        const float* sk = kp + (size_t)n * HP3 + h * 12 + p * 3;
        const float* sv = vp + (size_t)n * HP3 + h * 12 + p * 3;
#pragma unroll
        for (int e = 0; e < 3; ++e) {
            float pq = sq[0] * R[e] + sq[1] * R[3 + e] + sq[2] * R[6 + e] + t[e];
            float pk = sk[0] * R[e] + sk[1] * R[3 + e] + sk[2] * R[6 + e] + t[e];
            float pv = sv[0] * R[e] + sv[1] * R[3 + e] + sv[2] * R[6 + e] + t[e];
            qpt[p * 3 + e] = pq;
            kpt[p * 3 + e] = pk;
            vpt[p * 3 + e] = pv;
            q2 += pq * pq;
            k2 += pk * pk;
        }
    }
    // qfeat (plain f32, WMMA-A staged via LDS in attn kernel)
#pragma unroll
    for (int f = 0; f < FEAT; ++f) {
        float val = (f < 32) ? SCALE * q[(size_t)n * CDIM + h * CH + f]
                  : (f < 44) ? SCALE * qpt[f - 32] : 0.0f;
        qf[f] = val;
    }
    // kfragB: feature f lives in fragment f>>5, chunk offset c=f&31
    {
        int kt = n >> 4, ki = n & 15;
#pragma unroll
        for (int f = 0; f < FEAT; ++f) {
            float val = (f < 32) ? k[(size_t)n * CDIM + h * CH + f]
                      : (f < 44) ? kpt[f - 32] : 0.0f;
            int frag = f >> 5, c = f & 31;
            int lane = ki + 16 * inv_half(c);
            kfragB[((((size_t)h * 64 + kt) * 2 + frag) * 32 + lane) * 16 + inv_j(c)]
                = f2bf(val);
        }
    }
    // vfragB: key n is reduction row; col (0..63) is output feature
    {
        int kc = n >> 5, c = n & 31;
        int half = inv_half(c), j = inv_j(c);
#pragma unroll
        for (int col = 0; col < FEAT; ++col) {
            float val = (col < 32) ? v[(size_t)n * CDIM + h * CH + col]
                      : (col < 44) ? vpt[col - 32] : 0.0f;
            int ct = col >> 4;
            int lane = (col & 15) + 16 * half;
            vfragB[((((size_t)h * 32 + kc) * 4 + ct) * 32 + lane) * 16 + j]
                = f2bf(val);
        }
    }
    rq[(size_t)h * NTOK + n] = -0.5f * SCALE * q2;
    rk[(size_t)h * NTOK + n] = -0.5f * SCALE * k2;
}

// ---------------------------------------------------------------------------
// pair_bias: pbias_t[h][q][k] = pair[q,k,:] @ wpb[:,h] + bpb[h]
// One bandwidth-bound pass over 512 MB of `pair`, staged to LDS with CDNA5
// async-to-LDS loads (32 b128 in flight per wave); D tiles transposed via LDS
// so the 64 MB output is written in contiguous per-head runs.
// ---------------------------------------------------------------------------
__global__ void pair_bias_kernel(const float* __restrict__ pair,
                                 const float* __restrict__ wpb,
                                 const float* __restrict__ bpb,
                                 float* __restrict__ pbias_t) {
    extern __shared__ float smem[];
    float* sP = smem;                  // PBROWS * 128 f32 (128 KB)
    float* sW = sP + PBROWS * 128;     // 128 * 16
    float* sBias = sW + 128 * 16;      // 16
    float* sD = sBias + 16;            // PBROWS * 16 transpose buffer
    int tid = threadIdx.x;
    long long m0 = (long long)blockIdx.x * PBROWS;
    const float* gsrc = pair + m0 * DPAIR;
    for (int t = tid; t < PBROWS * 32; t += 256)
        async_ld_b128(sP + t * 4, gsrc + t * 4);
    for (int t = tid; t < 128 * 16; t += 256) {
        int kr = t >> 4, hc = t & 15;
        sW[t] = (hc < HEADS) ? wpb[kr * HEADS + hc] : 0.0f;
    }
    if (tid < 16) sBias[tid] = (tid < HEADS) ? bpb[tid] : 0.0f;
    wait_async0();
    __syncthreads();

    int wid = tid >> 5, lane = tid & 31;
    int row = lane & 15;
    int n = lane & 15, mb = (lane & 16) ? 8 : 0;
    for (int mt = wid; mt < PBROWS / 16; mt += 8) {
        int mrow = mt * 16;
        v8f c = {};
#pragma unroll
        for (int kc = 0; kc < 4; ++kc) {
            v16bf a, b;
#pragma unroll
            for (int j = 0; j < 16; ++j) {
                int ko = kc * 32 + frag_koff(lane, j);
                a[j] = f2bf(sP[(mrow + row) * DPAIR + ko]);
                b[j] = f2bf(sW[ko * 16 + row]);
            }
            c = __builtin_amdgcn_wmma_f32_16x16x32_bf16(false, a, false, b,
                                                        (short)0, c, false, false);
        }
#pragma unroll
        for (int r = 0; r < 8; ++r)
            sD[(mrow + mb + r) * 16 + n] = c[r] + sBias[n];
    }
    __syncthreads();
    // coalesced write-out: per head plane, PBROWS consecutive flattened (q,k)
    for (int t = tid; t < PBROWS * 16; t += 256) {
        int hh = t / PBROWS, mloc = t % PBROWS;
        pbias_t[(size_t)hh * NTOK * NTOK + m0 + mloc] = sD[mloc * 16 + hh];
    }
}

// ---------------------------------------------------------------------------
// Attention: one block per (16-query tile, head). 256 threads (8 waves).
// Phase 1: logits[16][1024] in LDS. Phase 2: softmax -> bf16 attention
// weights written once in WMMA-A-fragment layout. Phase 3: attn @ vfeat.
// ---------------------------------------------------------------------------
__global__ void attn_kernel(const float* __restrict__ qfeat,
                            const __bf16* __restrict__ kfragB,
                            const __bf16* __restrict__ vfragB,
                            const float* __restrict__ rq,
                            const float* __restrict__ rk,
                            const float* __restrict__ pbias_t,
                            float* __restrict__ asc,
                            float* __restrict__ apt) {
    extern __shared__ float smem[];
    float* lg        = smem;                    // 16*1024 raw logits
    float* qf        = lg + 16 * 1024;          // 16*64
    float* red       = qf + 16 * FEAT;          // 16*16
    float* rowsum    = red + 16 * 16;           // 16
    float* rqs       = rowsum + 16;             // 16
    float* pacc      = rqs + 16;                // 1024
    unsigned* abuf   = (unsigned*)(pacc + 1024);// 32 kc * 32 lanes * 8 dwords (32 KB)

    int h = blockIdx.y;
    int q0 = blockIdx.x * 16;
    int tid = threadIdx.x, wid = tid >> 5, lane = tid & 31;
    int row = lane & 15, nidx = lane & 15, mb = (lane & 16) ? 8 : 0;

    for (int t = tid; t < 16 * FEAT; t += 256)
        qf[t] = qfeat[((size_t)h * NTOK + q0 + (t >> 6)) * FEAT + (t & 63)];
    if (tid < 16) rqs[tid] = rq[(size_t)h * NTOK + q0 + tid];
    __syncthreads();

    // A fragments for the query tile (K = 64 -> 2 fragments), reused all k-tiles
    v16bf a0, a1;
#pragma unroll
    for (int j = 0; j < 16; ++j) {
        int ko = frag_koff(lane, j);
        a0[j] = f2bf(qf[row * FEAT + ko]);
        a1[j] = f2bf(qf[row * FEAT + 32 + ko]);
    }

    // -------- logits --------
    const v16bf* kfB = (const v16bf*)kfragB;
    const float* pbh = pbias_t + (size_t)h * NTOK * NTOK;
    for (int kt = wid; kt < NTOK / 16; kt += 8) {
        int k0 = kt * 16;
        size_t base = (((size_t)h * 64 + kt) * 2) * 32 + lane;
        v16bf b0 = kfB[base];
        v16bf b1 = kfB[base + 32];
        v8f c = {};
        c = __builtin_amdgcn_wmma_f32_16x16x32_bf16(false, a0, false, b0,
                                                    (short)0, c, false, false);
        c = __builtin_amdgcn_wmma_f32_16x16x32_bf16(false, a1, false, b1,
                                                    (short)0, c, false, false);
        float rkv = rk[(size_t)h * NTOK + k0 + nidx];
#pragma unroll
        for (int r = 0; r < 8; ++r) {
            int m = mb + r;
            float lv = c[r] + rqs[m] + rkv
                     + pbh[(size_t)(q0 + m) * NTOK + k0 + nidx];
            lg[m * NTOK + k0 + nidx] = lv;
        }
    }
    __syncthreads();

    // -------- softmax: max, then exp written once as bf16 A-fragments -------
    {
        int rrow = tid >> 4, rcol = tid & 15;
        float mx = -1e30f;
        for (int i = rcol; i < NTOK; i += 16) mx = fmaxf(mx, lg[rrow * NTOK + i]);
        red[rrow * 16 + rcol] = mx;
        __syncthreads();
        if (rcol == 0) {
            float m2 = red[rrow * 16];
            for (int i = 1; i < 16; ++i) m2 = fmaxf(m2, red[rrow * 16 + i]);
            red[rrow * 16] = m2;
        }
        __syncthreads();
        float rmax = red[rrow * 16];
        // this thread covers k = 2*rcol + 32*i; fragment slot fixed per thread
        int c = 2 * rcol;
        int alane = rrow + 16 * inv_half(c);
        int adw = alane * 8 + (inv_j(c) >> 1);
        float s = 0.0f;
        for (int i = 0; i < 32; ++i) {
            int kk = c + 32 * i;
            float e0 = __expf(lg[rrow * NTOK + kk] - rmax);
            float e1 = __expf(lg[rrow * NTOK + kk + 1] - rmax);
            s += e0 + e1;
            v2bf pk;
            pk[0] = f2bf(e0);
            pk[1] = f2bf(e1);
            abuf[i * 256 + adw] = __builtin_bit_cast(unsigned, pk);
        }
        __syncthreads();
        red[rrow * 16 + rcol] = s;
        __syncthreads();
        if (rcol == 0) {
            float t2 = 0.0f;
            for (int i = 0; i < 16; ++i) t2 += red[rrow * 16 + i];
            rowsum[rrow] = t2;
        }
        __syncthreads();
    }

    // -------- attn @ vfeat --------
    int ct = wid & 3;      // output column tile (0..3) of the 64-wide features
    int kh = wid >> 2;     // k-dimension half (0/1)
    const v16bf* vfB = (const v16bf*)vfragB;
    v8f o = {};
    for (int it = 0; it < 16; ++it) {
        int kc = kh * 16 + it;
        v16bf a = *(const v16bf*)(abuf + (size_t)kc * 256 + lane * 8);
        v16bf b = vfB[(((size_t)h * 32 + kc) * 4 + ct) * 32 + lane];
        o = __builtin_amdgcn_wmma_f32_16x16x32_bf16(false, a, false, b,
                                                    (short)0, o, false, false);
    }
    if (kh == 1) {
#pragma unroll
        for (int r = 0; r < 8; ++r) pacc[(ct * 32 + lane) * 8 + r] = o[r];
    }
    __syncthreads();
    if (kh == 0) {
#pragma unroll
        for (int r = 0; r < 8; ++r) o[r] += pacc[(ct * 32 + lane) * 8 + r];
#pragma unroll
        for (int r = 0; r < 8; ++r) {
            int m = mb + r;
            int col = ct * 16 + nidx;
            float val = o[r] / rowsum[m];
            if (col < 32)
                asc[(size_t)(q0 + m) * CDIM + h * CH + col] = val;
            else if (col < 44)
                apt[(size_t)(q0 + m) * HP3 + h * 12 + (col - 32)] = val;
        }
    }
}

// ---------------------------------------------------------------------------
// Residual + LayerNorm: one 128-thread block per token row.
// ---------------------------------------------------------------------------
__global__ void out_ln_kernel(const float* __restrict__ single,
                              const float* __restrict__ o1,
                              const float* __restrict__ o2,
                              const float* __restrict__ gamma,
                              const float* __restrict__ beta,
                              float* __restrict__ out) {
    __shared__ float xs[CDIM];
    __shared__ float rbuf[128];
    int n = blockIdx.x, tid = threadIdx.x;
    float loc = 0.0f;
    for (int c = tid; c < CDIM; c += 128) {
        float x = single[(size_t)n * CDIM + c] + o1[(size_t)n * CDIM + c]
                + o2[(size_t)n * CDIM + c];
        xs[c] = x;
        loc += x;
    }
    rbuf[tid] = loc;
    __syncthreads();
    for (int s = 64; s > 0; s >>= 1) {
        if (tid < s) rbuf[tid] += rbuf[tid + s];
        __syncthreads();
    }
    float mu = rbuf[0] / (float)CDIM;
    __syncthreads();
    float lv = 0.0f;
    for (int c = tid; c < CDIM; c += 128) {
        float d = xs[c] - mu;
        lv += d * d;
    }
    rbuf[tid] = lv;
    __syncthreads();
    for (int s = 64; s > 0; s >>= 1) {
        if (tid < s) rbuf[tid] += rbuf[tid + s];
        __syncthreads();
    }
    float inv = rsqrtf(rbuf[0] / (float)CDIM + EPS);
    for (int c = tid; c < CDIM; c += 128)
        out[(size_t)n * CDIM + c] = (xs[c] - mu) * inv * gamma[c] + beta[c];
}

// ---------------------------------------------------------------------------
extern "C" void kernel_launch(void* const* d_in, const int* in_sizes, int n_in,
                              void* d_out, int out_size, void* d_ws, size_t ws_size,
                              hipStream_t stream) {
    const float* single = (const float*)d_in[0];
    const float* pair   = (const float*)d_in[1];
    const float* rot    = (const float*)d_in[2];
    const float* trans  = (const float*)d_in[3];
    const float* wq     = (const float*)d_in[4];
    const float* bq     = (const float*)d_in[5];
    const float* wk     = (const float*)d_in[6];
    const float* bk     = (const float*)d_in[7];
    const float* wv     = (const float*)d_in[8];
    const float* bv     = (const float*)d_in[9];
    const float* wpb    = (const float*)d_in[10];
    const float* bpb    = (const float*)d_in[11];
    const float* wqp    = (const float*)d_in[12];
    const float* bqp    = (const float*)d_in[13];
    const float* wkp    = (const float*)d_in[14];
    const float* bkp    = (const float*)d_in[15];
    const float* wvp    = (const float*)d_in[16];
    const float* bvp    = (const float*)d_in[17];
    const float* wout   = (const float*)d_in[18];
    const float* bout   = (const float*)d_in[19];
    const float* wpo    = (const float*)d_in[20];
    const float* bpo    = (const float*)d_in[21];
    const float* gamma  = (const float*)d_in[22];
    const float* beta   = (const float*)d_in[23];
    float* out = (float*)d_out;

    float* wsf = (float*)d_ws;
    size_t off = 0;
    auto alloc = [&](size_t nf) { float* p = wsf + off; off += nf; return p; };
    float* q      = alloc((size_t)NTOK * CDIM);
    float* k      = alloc((size_t)NTOK * CDIM);
    float* v      = alloc((size_t)NTOK * CDIM);
    float* qp     = alloc((size_t)NTOK * HP3);
    float* kp     = alloc((size_t)NTOK * HP3);
    float* vp     = alloc((size_t)NTOK * HP3);
    float* qfeat  = alloc((size_t)HEADS * NTOK * FEAT);
    __bf16* kfragB = (__bf16*)alloc((size_t)HEADS * NTOK * FEAT / 2);
    __bf16* vfragB = (__bf16*)alloc((size_t)HEADS * NTOK * FEAT / 2);
    float* rqv    = alloc((size_t)HEADS * NTOK);
    float* rkv    = alloc((size_t)HEADS * NTOK);
    float* asc    = alloc((size_t)NTOK * CDIM);
    float* apt    = alloc((size_t)NTOK * HP3);
    float* o1     = alloc((size_t)NTOK * CDIM);
    float* o2     = alloc((size_t)NTOK * CDIM);
    float* pbias  = alloc((size_t)NTOK * NTOK * 16);

    // 1) scalar + point projections (6 GEMMs)
    dim3 gC(NTOK / 16, CDIM / 16);   // 64 x 24
    dim3 gP(NTOK / 16, HP3 / 16);    // 64 x 9
    gemm16_kernel<<<gC, 32, 0, stream>>>(single, wq, bq, q, NTOK, CDIM, CDIM);
    gemm16_kernel<<<gC, 32, 0, stream>>>(single, wk, bk, k, NTOK, CDIM, CDIM);
    gemm16_kernel<<<gC, 32, 0, stream>>>(single, wv, bv, v, NTOK, CDIM, CDIM);
    gemm16_kernel<<<gP, 32, 0, stream>>>(single, wqp, bqp, qp, NTOK, CDIM, HP3);
    gemm16_kernel<<<gP, 32, 0, stream>>>(single, wkp, bkp, kp, NTOK, CDIM, HP3);
    gemm16_kernel<<<gP, 32, 0, stream>>>(single, wvp, bvp, vp, NTOK, CDIM, HP3);

    // 2) frames + fragment-packed feature construction
    points_kernel<<<(NTOK * HEADS + 255) / 256, 256, 0, stream>>>(
        q, k, v, qp, kp, vp, rot, trans, qfeat, kfragB, vfragB, rqv, rkv);

    // 3) pair bias, single bandwidth-bound pass over `pair`
    size_t pbLds = (PBROWS * 128 + 128 * 16 + 16 + PBROWS * 16) * sizeof(float);
    pair_bias_kernel<<<(NTOK * NTOK) / PBROWS, 256, pbLds, stream>>>(pair, wpb, bpb, pbias);

    // 4) fused IPA attention
    size_t atLds = (16 * 1024 + 16 * FEAT + 16 * 16 + 16 + 16 + 1024
                    + 32 * 32 * 8) * sizeof(float);
    attn_kernel<<<dim3(NTOK / 16, HEADS), 256, atLds, stream>>>(
        qfeat, kfragB, vfragB, rqv, rkv, pbias, asc, apt);

    // 5) output projections
    gemm16_kernel<<<gC, 32, 0, stream>>>(asc, wout, bout, o1, NTOK, CDIM, CDIM);
    gemm16_kernel<<<gC, 32, 0, stream>>>(apt, wpo, bpo, o2, NTOK, HP3, CDIM);

    // 6) residual + layernorm
    out_ln_kernel<<<NTOK, 128, 0, stream>>>(single, o1, o2, gamma, beta, out);
}
